// MemTransformerLM_42795054137418
// MI455X (gfx1250) — compile-verified, matrix operand order checked by reading, not run on hardware
//
#include <hip/hip_runtime.h>
#include <hip/hip_bf16.h>
#include <cstdint>

// ---------------------------------------------------------------------------
// Transformer-XL forward on MI455X (gfx1250, wave32, WMMA).
// Memory-bound workload (~1-1.5GB HBM traffic vs ~4.3 TFLOP): all GEMMs use
// v_wmma_f32_16x16x32_f16 with on-the-fly f32->f16 conversion; the GEMM A-tile
// is staged into LDS via the CDNA5 async copy engine
// (global_load_async_to_lds_b128 / ASYNCcnt) and double-buffered so the copy
// of tile k+32 overlaps the WMMA work on tile k; logits log-softmax is fused
// (never materialize the 1GB [512,16,32001] tensor); hids are written directly
// into the new_mems output region.
// ---------------------------------------------------------------------------

typedef __attribute__((ext_vector_type(16))) _Float16 v16h;
typedef __attribute__((ext_vector_type(8)))  float    v8f;

#define TGT_   512
#define MEM_   512
#define BSZ_   16
#define DM_    1024
#define NH_    16
#define DH_    64
#define DI_    4096
#define NL_    8
#define NTOK_  32001
#define KLEN_  1024

static __device__ __forceinline__ v8f wmma16(v16h a, v16h b, v8f c) {
  // D = A(16x32 f16) x B(32x16 f16)^T-layout + C(16x16 f32)
  return __builtin_amdgcn_wmma_f32_16x16x32_f16(false, a, false, b, (short)0, c,
                                                false, false);
}

// A/B fragment (16x32 f16, one matrix row per lane, K striped per ISA 7.12.2):
// lane l: row = l&15, kbase = (l>>4)*8; halves 0..7 -> K kbase..kbase+7,
// halves 8..15 -> K kbase+16..kbase+23. Caller pre-offsets pointer by kbase.
static __device__ __forceinline__ v16h frag_load(const float* __restrict__ p) {
  const float4* q = (const float4*)p;
  float4 x0 = q[0], x1 = q[1], x2 = q[4], x3 = q[5];
  v16h f;
  f[0]  = (_Float16)x0.x; f[1]  = (_Float16)x0.y; f[2]  = (_Float16)x0.z; f[3]  = (_Float16)x0.w;
  f[4]  = (_Float16)x1.x; f[5]  = (_Float16)x1.y; f[6]  = (_Float16)x1.z; f[7]  = (_Float16)x1.w;
  f[8]  = (_Float16)x2.x; f[9]  = (_Float16)x2.y; f[10] = (_Float16)x2.z; f[11] = (_Float16)x2.w;
  f[12] = (_Float16)x3.x; f[13] = (_Float16)x3.y; f[14] = (_Float16)x3.z; f[15] = (_Float16)x3.w;
  return f;
}

static __device__ __forceinline__ v16h frag_load_bias(const float* __restrict__ p,
                                                      const float* __restrict__ bias) {
  v16h f;
#pragma unroll
  for (int h = 0; h < 8; ++h) f[h]     = (_Float16)(p[h]      + bias[h]);
#pragma unroll
  for (int h = 0; h < 8; ++h) f[8 + h] = (_Float16)(p[16 + h] + bias[16 + h]);
  return f;
}

// C/D fragment store: VGPR r, lanes 0-15 -> row r, lanes 16-31 -> row r+8.
static __device__ __forceinline__ void store_tile(float* __restrict__ C, size_t ldc,
                                                  int row, int col, v8f acc,
                                                  float bv, int relu) {
#pragma unroll
  for (int r = 0; r < 8; ++r) {
    float v = acc[r] + bv;
    if (relu) v = fmaxf(v, 0.0f);
    C[(size_t)(row + r) * ldc + col] = v;
  }
}

// CDNA5 async copy: per-lane 16B global->LDS, tracked by ASYNCcnt.
static __device__ __forceinline__ void async_copy_b128(unsigned ldsAddr,
                                                       unsigned gOff,
                                                       unsigned long long gBase) {
  asm volatile("global_load_async_to_lds_b128 %0, %1, %2"
               :: "v"(ldsAddr), "v"(gOff), "s"(gBase)
               : "memory");
}
static __device__ __forceinline__ void wait_async0() {
  asm volatile("s_wait_asynccnt 0x0" ::: "memory");
}

// ---------------------------------------------------------------------------
// Generic GEMM: C[M,N] = A[M,K] @ B[N,K]^T (+bias, +relu). 256 threads =
// 8 waves, each wave owns a 32x32 tile (2x2 WMMA fragments).
// The shared 32-row A tile is staged in LDS with double-buffered async copies
// (each wave stages 4 rows = one b128 per lane); B streams from global (every
// weight matrix fits the 192MB L2, so cross-block B re-reads are L2 hits).
// Grid: (M/32, N/256). M%32==0, N%256==0, K%32==0 for every call below.
// ---------------------------------------------------------------------------
__global__ void gemm_nt_kernel(const float* __restrict__ A, int lda,
                               const float* __restrict__ B, int ldb,
                               float* __restrict__ C, int ldc, int K,
                               const float* __restrict__ bias, int relu) {
  __shared__ __align__(16) float sA[2][32 * 32];
  int lane = threadIdx.x & 31, wave = threadIdx.x >> 5;
  int row0 = blockIdx.x * 32;
  int col0 = blockIdx.y * 256 + wave * 32;
  int ar = lane & 15, kb = (lane >> 4) * 8;

  // async staging map: wave w stages rows [4w, 4w+4); lane -> (row, 4 floats)
  int crow = wave * 4 + (lane >> 3);
  int ccol = (lane & 7) * 4;
  unsigned lds0 = (unsigned)(uintptr_t)&sA[0][crow * 32 + ccol];
  unsigned lds1 = (unsigned)(uintptr_t)&sA[1][crow * 32 + ccol];
  unsigned long long gBase = (unsigned long long)(uintptr_t)A;
  unsigned gOff = (unsigned)(((size_t)(row0 + crow) * lda + ccol) * sizeof(float));

  const float* Bp0 = B + (size_t)(col0 + ar) * ldb + kb;
  const float* Bp1 = B + (size_t)(col0 + 16 + ar) * ldb + kb;

  async_copy_b128(lds0, gOff, gBase);  // prologue: stage k=0 into buffer 0

  v8f a00 = {}, a01 = {}, a10 = {}, a11 = {};
  int cur = 0;
  for (int k = 0; k < K; k += 32) {
    wait_async0();     // this wave's slice of tile(cur) landed in LDS
    __syncthreads();   // all slices visible; all waves done reading prev buf
    if (k + 32 < K) {  // overlap copy of tile(k+32) with compute on tile(k)
      async_copy_b128(cur ? lds0 : lds1,
                      gOff + (unsigned)((k + 32) * sizeof(float)), gBase);
    }
    const float* sa = sA[cur];
    __builtin_prefetch(Bp0 + k + 256, 0, 1);
    __builtin_prefetch(Bp1 + k + 256, 0, 1);
    v16h fa0 = frag_load(sa + ar * 32 + kb);          // ds_load path
    v16h fa1 = frag_load(sa + (16 + ar) * 32 + kb);
    v16h fb0 = frag_load(Bp0 + k);
    v16h fb1 = frag_load(Bp1 + k);
    a00 = wmma16(fa0, fb0, a00);
    a01 = wmma16(fa0, fb1, a01);
    a10 = wmma16(fa1, fb0, a10);
    a11 = wmma16(fa1, fb1, a11);
    cur ^= 1;
  }
  int cr = 8 * (lane >> 4), cc = lane & 15;
  float b0 = bias ? bias[col0 + cc] : 0.0f;
  float b1 = bias ? bias[col0 + 16 + cc] : 0.0f;
  store_tile(C, (size_t)ldc, row0 + cr,      col0 + cc,      a00, b0, relu);
  store_tile(C, (size_t)ldc, row0 + cr,      col0 + 16 + cc, a01, b1, relu);
  store_tile(C, (size_t)ldc, row0 + 16 + cr, col0 + cc,      a10, b0, relu);
  store_tile(C, (size_t)ldc, row0 + 16 + cr, col0 + 16 + cc, a11, b1, relu);
}

// ---------------------------------------------------------------------------
// Batched per-(b,head) score GEMM:
//   useRK=0: AC[i,j] = (q[i]+r_w_bias) . k[j]     (K rows from heads)
//   useRK=1: QR[i,r] = (q[i]+r_r_bias) . rk[r]    (rows from rk buffer)
// heads layout: [klen, bsz, 3*1024]; q = last 512 rows of cols [0,1024),
// k = cols [1024,2048). out: [bn, 512, 1024]. Grid (16, 4, 256), block 256.
// ---------------------------------------------------------------------------
__global__ void attn_qk_kernel(const float* __restrict__ heads,
                               const float* __restrict__ rkb,
                               const float* __restrict__ rbias,
                               float* __restrict__ out, int useRK) {
  int lane = threadIdx.x & 31, wave = threadIdx.x >> 5;
  int bn = blockIdx.z, b = bn >> 4, n = bn & 15;
  int i0 = blockIdx.x * 32;
  int j0 = blockIdx.y * 256 + wave * 32;
  int ar = lane & 15, kb = (lane >> 4) * 8;
  const float* bias = rbias + n * 64 + kb;
  const float* Ap0 = heads + ((size_t)(MEM_ + i0 + ar) * BSZ_ + b) * 3072 + n * 64 + kb;
  const float* Ap1 = Ap0 + (size_t)16 * BSZ_ * 3072;
  const float *Bp0, *Bp1;
  if (useRK) {
    Bp0 = rkb + (size_t)(j0 + ar) * DM_ + n * 64 + kb;
    Bp1 = Bp0 + (size_t)16 * DM_;
  } else {
    Bp0 = heads + ((size_t)(j0 + ar) * BSZ_ + b) * 3072 + 1024 + n * 64 + kb;
    Bp1 = Bp0 + (size_t)16 * BSZ_ * 3072;
  }
  v8f a00 = {}, a01 = {}, a10 = {}, a11 = {};
#pragma unroll
  for (int k = 0; k < 64; k += 32) {
    v16h fa0 = frag_load_bias(Ap0 + k, bias + k);
    v16h fa1 = frag_load_bias(Ap1 + k, bias + k);
    v16h fb0 = frag_load(Bp0 + k);
    v16h fb1 = frag_load(Bp1 + k);
    a00 = wmma16(fa0, fb0, a00);
    a01 = wmma16(fa0, fb1, a01);
    a10 = wmma16(fa1, fb0, a10);
    a11 = wmma16(fa1, fb1, a11);
  }
  float* Cb = out + (size_t)bn * TGT_ * KLEN_;
  int cr = 8 * (lane >> 4), cc = lane & 15;
  store_tile(Cb, KLEN_, i0 + cr,      j0 + cc,      a00, 0.f, 0);
  store_tile(Cb, KLEN_, i0 + cr,      j0 + 16 + cc, a01, 0.f, 0);
  store_tile(Cb, KLEN_, i0 + 16 + cr, j0 + cc,      a10, 0.f, 0);
  store_tile(Cb, KLEN_, i0 + 16 + cr, j0 + 16 + cc, a11, 0.f, 0);
}

// ---------------------------------------------------------------------------
// prob @ V per (b,head): vec[i,d] = sum_j P[bn,i,j] * vh[j,b,n,d].
// Grid (16, 1, 256), block 64 (2 waves, each 32 of the 64 head dims).
// ---------------------------------------------------------------------------
__global__ void attn_av_kernel(const float* __restrict__ P,
                               const float* __restrict__ heads,
                               float* __restrict__ vec) {
  int lane = threadIdx.x & 31, wave = threadIdx.x >> 5;
  int bn = blockIdx.z, b = bn >> 4, n = bn & 15;
  int i0 = blockIdx.x * 32;
  int col0 = wave * 32;
  int ar = lane & 15, kb = (lane >> 4) * 8;
  const float* Ap0 = P + (size_t)bn * TGT_ * KLEN_ + (size_t)(i0 + ar) * KLEN_ + kb;
  const float* Ap1 = Ap0 + (size_t)16 * KLEN_;
  int d0 = col0 + ar, d1 = d0 + 16;
  const float* vbase = heads + (size_t)b * 3072 + 2048 + n * 64;
  const size_t js = (size_t)BSZ_ * 3072;
  v8f a00 = {}, a01 = {}, a10 = {}, a11 = {};
  for (int k = 0; k < KLEN_; k += 32) {
    v16h fa0 = frag_load(Ap0 + k);
    v16h fa1 = frag_load(Ap1 + k);
    v16h fb0, fb1;
#pragma unroll
    for (int h = 0; h < 8; ++h) {
      int j = k + kb + h;
      fb0[h]     = (_Float16)vbase[(size_t)j * js + d0];
      fb1[h]     = (_Float16)vbase[(size_t)j * js + d1];
      fb0[8 + h] = (_Float16)vbase[(size_t)(j + 16) * js + d0];
      fb1[8 + h] = (_Float16)vbase[(size_t)(j + 16) * js + d1];
    }
    a00 = wmma16(fa0, fb0, a00);
    a01 = wmma16(fa0, fb1, a01);
    a10 = wmma16(fa1, fb0, a10);
    a11 = wmma16(fa1, fb1, a11);
  }
  int cr = 8 * (lane >> 4), cc = lane & 15;
#pragma unroll
  for (int fr = 0; fr < 2; ++fr) {
    v8f ac0 = fr ? a10 : a00;
    v8f ac1 = fr ? a11 : a01;
#pragma unroll
    for (int r = 0; r < 8; ++r) {
      int row = i0 + fr * 16 + cr + r;
      size_t base = ((size_t)row * BSZ_ + b) * DM_ + n * 64;
      vec[base + col0 + cc]      = ac0[r];
      vec[base + col0 + 16 + cc] = ac1[r];
    }
  }
}

// ---------------------------------------------------------------------------
// score = mask ? -1e30 : (AC + rel_shift(QR)) * 1/sqrt(64)
// rel_shift(x)[i,j] = x[i, j + qlen-1 - i]; mask: j > i + mlen.
// ---------------------------------------------------------------------------
__global__ void combine_kernel(float* __restrict__ S, const float* __restrict__ QR) {
  int e = blockIdx.x * 256 + threadIdx.x;  // < 256*512*1024
  int rem = e & ((1 << 19) - 1);
  int i = rem >> 10, j = rem & 1023;
  if (j > i + MEM_) {
    S[e] = -1e30f;
  } else {
    S[e] = 0.125f * (S[e] + QR[(size_t)(e - j) + (j + TGT_ - 1 - i)]);
  }
}

// Softmax over klen=1024 per (bn, i). Grid (512, 256), block 256.
__global__ void softmax_kernel(float* __restrict__ S) {
  size_t row = (size_t)blockIdx.y * TGT_ + blockIdx.x;
  float* p = S + row * KLEN_;
  int tid = threadIdx.x;
  float v[4], m = -3.4e38f;
#pragma unroll
  for (int k = 0; k < 4; ++k) { v[k] = p[tid + k * 256]; m = fmaxf(m, v[k]); }
  __shared__ float red[256];
  red[tid] = m; __syncthreads();
  for (int off = 128; off; off >>= 1) {
    if (tid < off) red[tid] = fmaxf(red[tid], red[tid + off]);
    __syncthreads();
  }
  m = red[0]; __syncthreads();
  float s = 0.f;
#pragma unroll
  for (int k = 0; k < 4; ++k) { v[k] = __expf(v[k] - m); s += v[k]; }
  red[tid] = s; __syncthreads();
  for (int off = 128; off; off >>= 1) {
    if (tid < off) red[tid] += red[tid + off];
    __syncthreads();
  }
  float inv = 1.0f / red[0];
#pragma unroll
  for (int k = 0; k < 4; ++k) p[tid + k * 256] = v[k] * inv;
}

// out = LayerNorm(x + r) * g + b ; one block per row of 1024.
__global__ void ln_res_kernel(const float* __restrict__ x, const float* __restrict__ r,
                              const float* __restrict__ g, const float* __restrict__ be,
                              float* __restrict__ out) {
  int row = blockIdx.x, tid = threadIdx.x;
  const float* xp = x + (size_t)row * DM_;
  const float* rp = r + (size_t)row * DM_;
  float v[4], s = 0.f, ss = 0.f;
#pragma unroll
  for (int k = 0; k < 4; ++k) {
    int d = tid + k * 256;
    float t = xp[d] + rp[d];
    v[k] = t; s += t; ss += t * t;
  }
  __shared__ float rs[256], rq[256];
  rs[tid] = s; rq[tid] = ss; __syncthreads();
  for (int off = 128; off; off >>= 1) {
    if (tid < off) { rs[tid] += rs[tid + off]; rq[tid] += rq[tid + off]; }
    __syncthreads();
  }
  float mean = rs[0] * (1.0f / DM_);
  float var = rq[0] * (1.0f / DM_) - mean * mean;
  float rstd = rsqrtf(var + 1e-5f);
  float* op = out + (size_t)row * DM_;
#pragma unroll
  for (int k = 0; k < 4; ++k) {
    int d = tid + k * 256;
    op[d] = (v[k] - mean) * rstd * g[d] + be[d];
  }
}

// word_emb = emb_w[data+1] * sqrt(D_MODEL); one block per (t,b) row.
__global__ void embed_kernel(const int* __restrict__ data, const float* __restrict__ emb,
                             float* __restrict__ out) {
  int row = blockIdx.x, tid = threadIdx.x;
  int tok = data[row] + 1;
  const float* src = emb + (size_t)tok * DM_;
  float* dst = out + (size_t)row * DM_;
#pragma unroll
  for (int k = 0; k < 4; ++k) { int d = tid + k * 256; dst[d] = src[d] * 32.0f; }
}

// pos_emb[p,d]: pos = klen-1-p; [sin(pos*invf) | cos(pos*invf)]
__global__ void posemb_kernel(float* __restrict__ pos) {
  int idx = blockIdx.x * 256 + threadIdx.x;  // < 1024*1024
  int p = idx >> 10, d = idx & 1023;
  float ps = (float)(KLEN_ - 1 - p);
  int k2 = (d < 512) ? d : d - 512;
  float freq = __expf(-0.017988946e0f * (float)k2);  // 10000^(-2k/1024)
  float ang = ps * freq;
  pos[idx] = (d < 512) ? __sinf(ang) : __cosf(ang);
}

// cat = [mems_l ; core]  ([1024,16,1024])
__global__ void concat_kernel(const float* __restrict__ memsl,
                              const float* __restrict__ core,
                              float* __restrict__ cat) {
  int idx = blockIdx.x * 256 + threadIdx.x;  // < 16777216
  const int half = MEM_ * BSZ_ * DM_;
  cat[idx] = (idx < half) ? memsl[idx] : core[idx - half];
}

// Fused tied-embedding logits + log-softmax + NLL per (t,b) row.
__global__ void loss_kernel(const float* __restrict__ core, const float* __restrict__ emb,
                            const float* __restrict__ ob, const int* __restrict__ target,
                            float* __restrict__ loss) {
  int row = blockIdx.x, tid = threadIdx.x;
  __shared__ __align__(16) float c[DM_];
  for (int i = tid; i < DM_; i += 256) c[i] = core[(size_t)row * DM_ + i];
  __shared__ float sm[256], ssum[256], sxt;
  __syncthreads();
  int tgt = target[row] + 1;
  const float4* c4 = (const float4*)c;
  float m = -3.4e38f, s = 0.f;
  for (int v = tid; v < NTOK_; v += 256) {
    const float4* e4 = (const float4*)(emb + (size_t)v * DM_);
    float d = 0.f;
    for (int i = 0; i < DM_ / 4; ++i) {
      float4 a = e4[i], bb = c4[i];
      d = fmaf(a.x, bb.x, d); d = fmaf(a.y, bb.y, d);
      d = fmaf(a.z, bb.z, d); d = fmaf(a.w, bb.w, d);
    }
    float x = d + ob[v];
    if (v == tgt) sxt = x;
    float mn = fmaxf(m, x);
    s = s * __expf(m - mn) + __expf(x - mn);
    m = mn;
  }
  sm[tid] = m; ssum[tid] = s; __syncthreads();
  for (int off = 128; off; off >>= 1) {
    if (tid < off) {
      float m2 = sm[tid + off], s2 = ssum[tid + off];
      float mn = fmaxf(sm[tid], m2);
      ssum[tid] = ssum[tid] * __expf(sm[tid] - mn) + s2 * __expf(m2 - mn);
      sm[tid] = mn;
    }
    __syncthreads();
  }
  if (tid == 0) {
    float nll = -(sxt - sm[0] - __logf(ssum[0]));
    loss[row] = (tgt == 0) ? 0.0f : nll;
  }
}

// Zero new_mems where last input token is PAD (after offset).
__global__ void padmask_kernel(const int* __restrict__ data, float* __restrict__ hids) {
  int idx = blockIdx.x * 256 + threadIdx.x;  // < 9*512*16*1024
  int b = (idx >> 10) & 15;
  if (data[(TGT_ - 1) * BSZ_ + b] + 1 == 0) hids[idx] = 0.0f;
}

// ---------------------------------------------------------------------------
extern "C" void kernel_launch(void* const* d_in, const int* in_sizes, int n_in,
                              void* d_out, int out_size, void* d_ws, size_t ws_size,
                              hipStream_t stream) {
  const int*   data   = (const int*)d_in[0];
  const int*   target = (const int*)d_in[1];
  const float* mems   = (const float*)d_in[3];
  const float* emb_w  = (const float*)d_in[4];
  const float* out_b  = (const float*)d_in[5];
  const float* rwb    = (const float*)d_in[6];
  const float* rrb    = (const float*)d_in[7];
  const float* qkv_w  = (const float*)d_in[8];
  const float* rnet_w = (const float*)d_in[9];
  const float* o_w    = (const float*)d_in[10];
  const float* ln1s   = (const float*)d_in[11];
  const float* ln1b   = (const float*)d_in[12];
  const float* ffw1   = (const float*)d_in[13];
  const float* ffb1   = (const float*)d_in[14];
  const float* ffw2   = (const float*)d_in[15];
  const float* ffb2   = (const float*)d_in[16];
  const float* ln2s   = (const float*)d_in[17];
  const float* ln2b   = (const float*)d_in[18];

  float* loss = (float*)d_out;                       // [512,16]
  float* hids = (float*)d_out + TGT_ * BSZ_;         // [9,512,16,1024] == new_mems

  const size_t S_SZ   = 256UL * TGT_ * KLEN_;        // 134,217,728 floats
  const size_t H_SZ   = (size_t)KLEN_ * BSZ_ * 3072; // heads
  const size_t HID_SZ = (size_t)TGT_ * BSZ_ * DM_;   // per-layer hidden

  float* ws    = (float*)d_ws;
  float* S     = ws;                 // scores / probs
  float* QR    = S + S_SZ;           // shifted-BD source
  float* heads = QR + S_SZ;
  float* vec   = heads + H_SZ;
  float* tmp   = vec + HID_SZ;
  float* h2    = tmp + HID_SZ;
  float* pos   = h2 + HID_SZ;
  float* rkb   = pos + (size_t)KLEN_ * DM_;
  float* cat   = S;                  // alias: cat dead before S written
  float* ff1   = QR;                 // alias: QR dead before ff1 written

  embed_kernel<<<TGT_ * BSZ_, 256, 0, stream>>>(data, emb_w, hids);
  posemb_kernel<<<(KLEN_ * DM_) / 256, 256, 0, stream>>>(pos);

  for (int l = 0; l < NL_; ++l) {
    const float* mem_l = mems + (size_t)l * HID_SZ;
    float* core  = hids + (size_t)l * HID_SZ;
    float* coren = hids + (size_t)(l + 1) * HID_SZ;

    concat_kernel<<<(KLEN_ * BSZ_ * DM_) / 256, 256, 0, stream>>>(mem_l, core, cat);
    // heads = cat @ qkv_w^T   [16384,3072]
    gemm_nt_kernel<<<dim3(512, 12), 256, 0, stream>>>(
        cat, DM_, qkv_w + (size_t)l * 3072 * DM_, DM_, heads, 3072, DM_, nullptr, 0);
    // rk = pos_emb @ r_net_w^T   [1024,1024]
    gemm_nt_kernel<<<dim3(32, 4), 256, 0, stream>>>(
        pos, DM_, rnet_w + (size_t)l * DM_ * DM_, DM_, rkb, DM_, DM_, nullptr, 0);
    // AC and QR score matrices per (b,head)
    attn_qk_kernel<<<dim3(16, 4, 256), 256, 0, stream>>>(heads, nullptr, rwb, S, 0);
    attn_qk_kernel<<<dim3(16, 4, 256), 256, 0, stream>>>(heads, rkb, rrb, QR, 1);
    combine_kernel<<<(unsigned)(S_SZ / 256), 256, 0, stream>>>(S, QR);
    softmax_kernel<<<dim3(512, 256), 256, 0, stream>>>(S);
    attn_av_kernel<<<dim3(16, 1, 256), 64, 0, stream>>>(S, heads, vec);
    // attn out projection + post-LN
    gemm_nt_kernel<<<dim3(256, 4), 256, 0, stream>>>(
        vec, DM_, o_w + (size_t)l * DM_ * DM_, DM_, tmp, DM_, DM_, nullptr, 0);
    ln_res_kernel<<<TGT_ * BSZ_, 256, 0, stream>>>(core, tmp, ln1s + l * DM_,
                                                   ln1b + l * DM_, h2);
    // FFN
    gemm_nt_kernel<<<dim3(256, 16), 256, 0, stream>>>(
        h2, DM_, ffw1 + (size_t)l * DI_ * DM_, DM_, ff1, DI_, DM_, ffb1 + l * DI_, 1);
    gemm_nt_kernel<<<dim3(256, 4), 256, 0, stream>>>(
        ff1, DI_, ffw2 + (size_t)l * DM_ * DI_, DI_, tmp, DM_, DI_, ffb2 + l * DM_, 0);
    ln_res_kernel<<<TGT_ * BSZ_, 256, 0, stream>>>(h2, tmp, ln2s + l * DM_,
                                                   ln2b + l * DM_, coren);
  }

  // fused tied-embedding logits + log-softmax + NLL (before pad-masking mems)
  loss_kernel<<<TGT_ * BSZ_, 256, 0, stream>>>(hids + (size_t)NL_ * HID_SZ, emb_w,
                                               out_b, target, loss);
  padmask_kernel<<<(unsigned)(9UL * HID_SZ / 256), 256, 0, stream>>>(data, hids);
}
